// TransformerModel_85392539779695
// MI455X (gfx1250) — matrix operand order repaired and checked by reading, not compiled, not run on hardware
//
#include <hip/hip_runtime.h>

// ---------------------------------------------------------------------------
// Model constants (match reference)
// ---------------------------------------------------------------------------
#define Bc   2
#define Sc   1024
#define Dc   1024
#define Hc   16
#define DHc  64
#define DFFc 4096
#define Vc   32000
#define Lc   4
#define BSc  (Bc * Sc)          // 2048 rows in the token stream

typedef __bf16 bf16;
typedef __attribute__((ext_vector_type(16))) __bf16 v16bf;
typedef __attribute__((ext_vector_type(8)))  __bf16 v8bf;
typedef __attribute__((ext_vector_type(8)))  float  v8f;
typedef int v4i __attribute__((vector_size(16)));   // matches builtin prototype

// CDNA5 async global->LDS copy (ASYNCcnt-tracked); fallback = load + ds_store.
#if defined(__HIP_DEVICE_COMPILE__) && \
    __has_builtin(__builtin_amdgcn_global_load_async_to_lds_b128)
#define ASYNC_COPY 1
#define WAIT_ASYNC(n) asm volatile("s_wait_asynccnt %0" ::"i"(n) : "memory")
#else
#define WAIT_ASYNC(n)
#endif

__device__ __forceinline__ void async_copy16(const bf16* g, bf16* l)
{
#if defined(ASYNC_COPY)
    __builtin_amdgcn_global_load_async_to_lds_b128(
        (__attribute__((address_space(1))) v4i*)g,
        (__attribute__((address_space(3))) v4i*)l, 0, 0);
#else
    *(v8bf*)l = *(const v8bf*)g;
#endif
}

// ---------------------------------------------------------------------------
// Batched bf16 WMMA GEMM:  C[z] = alpha * A[z] x B[z]^T (+bias)(+res)(relu?)
//   A: [M,K] bf16 row-major (lda);      batch offset z*sA
//   B: [N,K] bf16 row-major (ldb);      batch offset (z/zdivB)*sB1+(z%zdivB)*sB2
//   C: fp32 (Cf) or bf16 (Ch);          batch offset (z/zdivC)*sC1+(z%zdivC)*sC2
//       element: row*ldc+cn, or cn*ldc+row when transC (used to store V^T)
//   bias: fp32 per output column, batch stride sBias
//   res : fp32 residual read at the same index as C
// Tile: 128x64 per workgroup (8 waves); each wave owns 32x32 = 4 WMMA tiles.
// Double-buffered LDS, async staged.
// ---------------------------------------------------------------------------
#define BM  128
#define BN  64
#define BK  32
#define LDP 40   // LDS row pitch in bf16 (32 + 8 pad; 80B = 5x16B)

__global__ __launch_bounds__(256) void gemm_bf16_wmma(
    const bf16* __restrict__ A, const bf16* __restrict__ Bm,
    float* __restrict__ Cf, bf16* __restrict__ Ch,
    const float* __restrict__ bias, const float* __restrict__ res,
    int K, int lda, int ldb, int ldc,
    long sA, long sB1, long sB2, int zdivB,
    long sC1, long sC2, int zdivC, long sBias,
    float alpha, int transC, int relu)
{
    __shared__ bf16 aLDS[2][BM][LDP];   // [m][k]
    __shared__ bf16 bLDS[2][BN][LDP];   // [n][k]

    const int tid  = threadIdx.x;
    const int lane = tid & 31;
    const int wave = tid >> 5;       // 0..7
    const int wm   = wave & 3;       // 4 strips of 32 rows
    const int wn   = wave >> 2;      // 2 strips of 32 cols

    const int z = blockIdx.z;
    A  += (long)z * sA;
    Bm += (long)(z / zdivB) * sB1 + (long)(z % zdivB) * sB2;
    const long cbase = (long)(z / zdivC) * sC1 + (long)(z % zdivC) * sC2;

    const int m0 = blockIdx.y * BM;
    const int n0 = blockIdx.x * BN;

    // staging coords: 256 threads x 16B vectors (A: 2 rows/thread, B: 1)
    const int ar = tid >> 2;             // 0..63
    const int ak = (tid & 3) * 8;        // 0,8,16,24

    // fragment coords (ISA 7.12.2 wave32 layouts)
    const int half = lane >> 4;
    const int l15  = lane & 15;
    const int mr0  = wm * 32 + l15;      // A-frag rows (mi*16)
    const int nr0  = wn * 32 + l15;      // B-frag cols (ni*16)

    v8f acc00 = {}, acc01 = {}, acc10 = {}, acc11 = {};

    const int nk = K / BK;
    // prologue: stage k-tile 0 into buffer 0
    {
        const bf16* ga = A + (long)(m0 + ar) * lda + ak;
        async_copy16(ga, &aLDS[0][ar][ak]);
        async_copy16(ga + 64L * lda, &aLDS[0][ar + 64][ak]);
        async_copy16(Bm + (long)(n0 + ar) * ldb + ak, &bLDS[0][ar][ak]);
    }

    for (int i = 0; i < nk; ++i) {
        const int cur = i & 1;
        __syncthreads();                       // prior reads of buf[cur^1] done
        if (i + 1 < nk) {
            const int k0 = (i + 1) * BK;
            const bf16* ga = A + (long)(m0 + ar) * lda + (k0 + ak);
            async_copy16(ga, &aLDS[cur ^ 1][ar][ak]);
            async_copy16(ga + 64L * lda, &aLDS[cur ^ 1][ar + 64][ak]);
            async_copy16(Bm + (long)(n0 + ar) * ldb + (k0 + ak),
                         &bLDS[cur ^ 1][ar][ak]);
            WAIT_ASYNC(3);                     // current tile landed; next 3 fly
        } else {
            WAIT_ASYNC(0);
        }
        __syncthreads();                       // current tile visible to all

        // A frag 16x32: lane<16 -> k {0..7,16..23}; lane>=16 -> +8
        union F { v16bf v; v8bf h[2]; } a0, a1, b0, b1;
        a0.h[0] = *(const v8bf*)&aLDS[cur][mr0][half * 8];
        a0.h[1] = *(const v8bf*)&aLDS[cur][mr0][16 + half * 8];
        a1.h[0] = *(const v8bf*)&aLDS[cur][mr0 + 16][half * 8];
        a1.h[1] = *(const v8bf*)&aLDS[cur][mr0 + 16][16 + half * 8];
        // B frag 32x16: lane<16 -> k {0..7,16..23}; lane>=16 -> +8
        b0.h[0] = *(const v8bf*)&bLDS[cur][nr0][half * 8];
        b0.h[1] = *(const v8bf*)&bLDS[cur][nr0][16 + half * 8];
        b1.h[0] = *(const v8bf*)&bLDS[cur][nr0 + 16][half * 8];
        b1.h[1] = *(const v8bf*)&bLDS[cur][nr0 + 16][16 + half * 8];

        acc00 = __builtin_amdgcn_wmma_f32_16x16x32_bf16(
                    false, a0.v, false, b0.v, (short)0, acc00, false, false);
        acc01 = __builtin_amdgcn_wmma_f32_16x16x32_bf16(
                    false, a0.v, false, b1.v, (short)0, acc01, false, false);
        acc10 = __builtin_amdgcn_wmma_f32_16x16x32_bf16(
                    false, a1.v, false, b0.v, (short)0, acc10, false, false);
        acc11 = __builtin_amdgcn_wmma_f32_16x16x32_bf16(
                    false, a1.v, false, b1.v, (short)0, acc11, false, false);
    }

    // epilogue — C frag: VGPR r, lanes 0-15: M=r, lanes 16-31: M=8+r, N=l15
    const int col = lane & 15;
#pragma unroll
    for (int mi = 0; mi < 2; ++mi) {
#pragma unroll
        for (int ni = 0; ni < 2; ++ni) {
            const v8f acc = (mi == 0) ? (ni == 0 ? acc00 : acc01)
                                      : (ni == 0 ? acc10 : acc11);
            const int cn = n0 + wn * 32 + ni * 16 + col;
#pragma unroll
            for (int r = 0; r < 8; ++r) {
                const int row = m0 + wm * 32 + mi * 16 + half * 8 + r;
                float v = acc[r] * alpha;
                if (bias) v += bias[(long)z * sBias + cn];
                const long cidx = cbase + (transC ? (long)cn * ldc + row
                                                  : (long)row * ldc + cn);
                if (res)  v += res[cidx];
                if (relu) v = v > 0.f ? v : 0.f;
                if (Cf) Cf[cidx] = v;
                else    Ch[cidx] = (bf16)v;
            }
        }
    }
}

// ---------------------------------------------------------------------------
// Elementwise / data-movement kernels
// ---------------------------------------------------------------------------
__global__ __launch_bounds__(256) void embed_pe_kernel(
    const int* __restrict__ idx, const float* __restrict__ emb,
    float* __restrict__ x)
{
    const int bs = blockIdx.x;
    const int s  = bs & (Sc - 1);
    const float* e = emb + (long)idx[bs] * Dc;
    float* xr = x + (long)bs * Dc;
    for (int c = threadIdx.x; c < Dc; c += 256) {
        const int j = c >> 1;
        const float freq = __expf((float)(2 * j) * (-9.210340371976184f / Dc));
        const float ang  = (float)s * freq;
        const float pe   = (c & 1) ? __cosf(ang) : __sinf(ang);
        xr[c] = e[c] * 32.0f + pe;        // sqrt(D)=32
    }
}

__global__ __launch_bounds__(256) void layernorm_bf16_kernel(
    const float* __restrict__ x, const float* __restrict__ g,
    const float* __restrict__ b, bf16* __restrict__ out)
{
    __shared__ float s1[256], s2[256];
    const int row = blockIdx.x;
    const float* xr = x + (long)row * Dc;
    float sum = 0.f, sq = 0.f;
    for (int c = threadIdx.x; c < Dc; c += 256) {
        const float v = xr[c]; sum += v; sq += v * v;
    }
    s1[threadIdx.x] = sum; s2[threadIdx.x] = sq;
    __syncthreads();
    for (int off = 128; off > 0; off >>= 1) {
        if (threadIdx.x < off) {
            s1[threadIdx.x] += s1[threadIdx.x + off];
            s2[threadIdx.x] += s2[threadIdx.x + off];
        }
        __syncthreads();
    }
    const float mean = s1[0] * (1.0f / Dc);
    const float var  = s2[0] * (1.0f / Dc) - mean * mean;
    const float rstd = rsqrtf(var + 1e-5f);
    bf16* orow = out + (long)row * Dc;
    for (int c = threadIdx.x; c < Dc; c += 256)
        orow[c] = (bf16)((xr[c] - mean) * rstd * g[c] + b[c]);
}

__global__ __launch_bounds__(256) void softmax_causal_kernel(
    const bf16* __restrict__ sc, bf16* __restrict__ pr)
{
    __shared__ float red[256];
    const int s = blockIdx.x;                 // query position
    const int z = blockIdx.y;                 // (h*B + b)
    const bf16* row = sc + ((long)z * Sc + s) * Sc;
    bf16*       out = pr + ((long)z * Sc + s) * Sc;
    const int n = s + 1;                      // causal length

    float mx = -1e30f;
    for (int t = threadIdx.x; t < n; t += 256)
        mx = fmaxf(mx, (float)row[t]);
    red[threadIdx.x] = mx; __syncthreads();
    for (int off = 128; off > 0; off >>= 1) {
        if (threadIdx.x < off)
            red[threadIdx.x] = fmaxf(red[threadIdx.x], red[threadIdx.x + off]);
        __syncthreads();
    }
    mx = red[0]; __syncthreads();

    float sum = 0.f;
    for (int t = threadIdx.x; t < n; t += 256)
        sum += __expf((float)row[t] - mx);
    red[threadIdx.x] = sum; __syncthreads();
    for (int off = 128; off > 0; off >>= 1) {
        if (threadIdx.x < off) red[threadIdx.x] += red[threadIdx.x + off];
        __syncthreads();
    }
    const float inv = 1.0f / red[0];

    for (int t = threadIdx.x; t < Sc; t += 256)
        out[t] = (t < n) ? (bf16)(__expf((float)row[t] - mx) * inv) : (bf16)0.0f;
}

__global__ __launch_bounds__(256) void f32_to_bf16_kernel(
    const float* __restrict__ in, bf16* __restrict__ out, long n)
{
    long i = (long)blockIdx.x * blockDim.x + threadIdx.x;
    const long stride = (long)gridDim.x * blockDim.x;
    for (; i < n; i += stride) out[i] = (bf16)in[i];
}

// in: [K,N] fp32 (+z*sIn)  ->  out: [N,K] bf16 (+z*sOut)
__global__ __launch_bounds__(256) void transpose_cvt_kernel(
    const float* __restrict__ in, bf16* __restrict__ out,
    int K, int N, long sIn, long sOut)
{
    __shared__ float t[32][33];
    const int z = blockIdx.z;
    in  += (long)z * sIn;
    out += (long)z * sOut;
    const int k0 = blockIdx.y * 32, n0 = blockIdx.x * 32;
    for (int i = threadIdx.y; i < 32; i += 8)
        t[i][threadIdx.x] = in[(long)(k0 + i) * N + n0 + threadIdx.x];
    __syncthreads();
    for (int i = threadIdx.y; i < 32; i += 8)
        out[(long)(n0 + i) * K + k0 + threadIdx.x] = (bf16)t[threadIdx.x][i];
}

// ---------------------------------------------------------------------------
// Host orchestration
// ---------------------------------------------------------------------------
static inline void launch_gemm(hipStream_t st, const bf16* A, const bf16* Bm,
                               float* Cf, bf16* Ch, const float* bias,
                               const float* res, int M, int N, int K,
                               int lda, int ldb, int ldc,
                               long sA, long sB1, long sB2, int zdivB,
                               long sC1, long sC2, int zdivC, long sBias,
                               float alpha, int transC, int relu, int batch)
{
    dim3 grid(N / BN, M / BM, batch), block(256);
    hipLaunchKernelGGL(gemm_bf16_wmma, grid, block, 0, st, A, Bm, Cf, Ch, bias,
                       res, K, lda, ldb, ldc, sA, sB1, sB2, zdivB,
                       sC1, sC2, zdivC, sBias, alpha, transC, relu);
}

static inline void launch_cvt(hipStream_t st, const float* in, bf16* out, long n)
{
    hipLaunchKernelGGL(f32_to_bf16_kernel, dim3(4096), dim3(256), 0, st, in, out, n);
}

static inline void launch_tcvt(hipStream_t st, const float* in, bf16* out,
                               int K, int N, long sIn, long sOut, int batch)
{
    dim3 grid(N / 32, K / 32, batch), block(32, 8);
    hipLaunchKernelGGL(transpose_cvt_kernel, grid, block, 0, st, in, out,
                       K, N, sIn, sOut);
}

extern "C" void kernel_launch(void* const* d_in, const int* in_sizes, int n_in,
                              void* d_out, int out_size, void* d_ws, size_t ws_size,
                              hipStream_t stream)
{
    (void)in_sizes; (void)n_in; (void)out_size; (void)ws_size;

    const int*   idx   = (const int*)  d_in[0];
    const float* emb   = (const float*)d_in[1];
    const float* cls_b = (const float*)d_in[2];
    const float* ln1_s = (const float*)d_in[3];
    const float* ln1_b = (const float*)d_in[4];
    const float* wq    = (const float*)d_in[5];
    const float* bq    = (const float*)d_in[6];
    const float* wk    = (const float*)d_in[7];
    const float* bk    = (const float*)d_in[8];
    const float* wv    = (const float*)d_in[9];
    const float* bv    = (const float*)d_in[10];
    const float* wo    = (const float*)d_in[11];
    const float* bo    = (const float*)d_in[12];
    const float* ln2_s = (const float*)d_in[13];
    const float* ln2_b = (const float*)d_in[14];
    const float* w1    = (const float*)d_in[15];
    const float* b1    = (const float*)d_in[16];
    const float* w2    = (const float*)d_in[17];
    const float* b2    = (const float*)d_in[18];
    const float* lnf_s = (const float*)d_in[19];
    const float* lnf_b = (const float*)d_in[20];
    float* logits = (float*)d_out;

    char* w = (char*)d_ws;
    auto carve = [&](size_t bytes) {
        char* p = w; w += (bytes + 255) & ~(size_t)255; return p;
    };
    float* xbuf  = (float*)carve((size_t)BSc * Dc * 4);        // residual stream
    bf16* hbuf   = (bf16*)carve((size_t)BSc * Dc * 2);         // LN output
    bf16* qbuf   = (bf16*)carve((size_t)BSc * Dc * 2);         // [H, B*S, DH]
    bf16* kbuf   = (bf16*)carve((size_t)BSc * Dc * 2);         // [H, B*S, DH]
    bf16* vtbuf  = (bf16*)carve((size_t)BSc * Dc * 2);         // [H, DH, B*S]
    bf16* scores = (bf16*)carve((size_t)Bc * Hc * Sc * Sc * 2);
    bf16* probs  = (bf16*)carve((size_t)Bc * Hc * Sc * Sc * 2);
    bf16* obuf   = (bf16*)carve((size_t)BSc * Dc * 2);         // [B,S,H*DH]
    bf16* ffmid  = (bf16*)carve((size_t)BSc * DFFc * 2);
    bf16* xnbf   = (bf16*)carve((size_t)BSc * Dc * 2);
    bf16* embbf  = (bf16*)carve((size_t)Vc * Dc * 2);          // [V,D] (= B^T)
    bf16* wqT    = (bf16*)carve((size_t)Hc * Dc * DHc * 2);    // [H, DH, D]
    bf16* wkT    = (bf16*)carve((size_t)Hc * Dc * DHc * 2);
    bf16* wvT    = (bf16*)carve((size_t)Hc * Dc * DHc * 2);
    bf16* woT    = (bf16*)carve((size_t)Dc * Dc * 2);          // [D, D]^T
    bf16* w1T    = (bf16*)carve((size_t)Dc * DFFc * 2);        // [DFF, D]
    bf16* w2T    = (bf16*)carve((size_t)DFFc * Dc * 2);        // [D, DFF]

    launch_cvt(stream, emb, embbf, (long)Vc * Dc);             // already [N,K]
    hipLaunchKernelGGL(embed_pe_kernel, dim3(BSc), dim3(256), 0, stream,
                       idx, emb, xbuf);

    const long WQKV = (long)Hc * Dc * DHc;
    for (int l = 0; l < Lc; ++l) {
        launch_tcvt(stream, wq + (long)l * WQKV, wqT, Dc, DHc,
                    (long)Dc * DHc, (long)DHc * Dc, Hc);
        launch_tcvt(stream, wk + (long)l * WQKV, wkT, Dc, DHc,
                    (long)Dc * DHc, (long)DHc * Dc, Hc);
        launch_tcvt(stream, wv + (long)l * WQKV, wvT, Dc, DHc,
                    (long)Dc * DHc, (long)DHc * Dc, Hc);
        launch_tcvt(stream, wo + (long)l * Dc * Dc, woT, Dc, Dc, 0, 0, 1);
        launch_tcvt(stream, w1 + (long)l * Dc * DFFc, w1T, Dc, DFFc, 0, 0, 1);
        launch_tcvt(stream, w2 + (long)l * DFFc * Dc, w2T, DFFc, Dc, 0, 0, 1);

        // h = LN1(x)
        hipLaunchKernelGGL(layernorm_bf16_kernel, dim3(BSc), dim3(256), 0,
                           stream, xbuf, ln1_s + l * Dc, ln1_b + l * Dc, hbuf);

        // Q,K: [H, B*S, DH];  V stored transposed: [H, DH, B*S]
        launch_gemm(stream, hbuf, wqT, nullptr, qbuf, bq + (long)l * Hc * DHc,
                    nullptr, BSc, DHc, Dc, Dc, Dc, DHc,
                    0, (long)DHc * Dc, 0, 1, (long)BSc * DHc, 0, 1, DHc,
                    1.0f, 0, 0, Hc);
        launch_gemm(stream, hbuf, wkT, nullptr, kbuf, bk + (long)l * Hc * DHc,
                    nullptr, BSc, DHc, Dc, Dc, Dc, DHc,
                    0, (long)DHc * Dc, 0, 1, (long)BSc * DHc, 0, 1, DHc,
                    1.0f, 0, 0, Hc);
        launch_gemm(stream, hbuf, wvT, nullptr, vtbuf, bv + (long)l * Hc * DHc,
                    nullptr, BSc, DHc, Dc, Dc, Dc, BSc,
                    0, (long)DHc * Dc, 0, 1, (long)DHc * BSc, 0, 1, DHc,
                    1.0f, 1 /*transC*/, 0, Hc);

        // scores = (Q K^T)/8;  batch z = h*B + b
        launch_gemm(stream, qbuf, kbuf, nullptr, scores, nullptr, nullptr,
                    Sc, Sc, DHc, DHc, DHc, Sc,
                    (long)Sc * DHc, (long)Sc * DHc, 0, 1,
                    (long)Sc * Sc, 0, 1, 0, 0.125f, 0, 0, Hc * Bc);

        hipLaunchKernelGGL(softmax_causal_kernel, dim3(Sc, Hc * Bc), dim3(256),
                           0, stream, scores, probs);

        // attn = P V  (B = V^T [DH, B*S]); heads land concatenated in [B,S,D]
        launch_gemm(stream, probs, vtbuf, nullptr, obuf, nullptr, nullptr,
                    Sc, DHc, Sc, Sc, BSc, Dc,
                    (long)Sc * Sc, (long)DHc * BSc, (long)Sc, Bc,
                    (long)DHc, (long)Sc * Dc, Bc, 0,
                    1.0f, 0, 0, Hc * Bc);

        // x = x + attn @ Wo + bo
        launch_gemm(stream, obuf, woT, xbuf, nullptr, bo + (long)l * Dc, xbuf,
                    BSc, Dc, Dc, Dc, Dc, Dc,
                    0, 0, 0, 1, 0, 0, 1, 0, 1.0f, 0, 0, 1);

        // h2 = LN2(x)
        hipLaunchKernelGGL(layernorm_bf16_kernel, dim3(BSc), dim3(256), 0,
                           stream, xbuf, ln2_s + l * Dc, ln2_b + l * Dc, hbuf);

        // ffmid = relu(h2 @ W1 + b1)
        launch_gemm(stream, hbuf, w1T, nullptr, ffmid, b1 + (long)l * DFFc,
                    nullptr, BSc, DFFc, Dc, Dc, Dc, DFFc,
                    0, 0, 0, 1, 0, 0, 1, 0, 1.0f, 0, 1, 1);

        // x = x + ffmid @ W2 + b2
        launch_gemm(stream, ffmid, w2T, xbuf, nullptr, b2 + (long)l * Dc, xbuf,
                    BSc, Dc, DFFc, DFFc, DFFc, Dc,
                    0, 0, 0, 1, 0, 0, 1, 0, 1.0f, 0, 0, 1);
    }

    hipLaunchKernelGGL(layernorm_bf16_kernel, dim3(BSc), dim3(256), 0, stream,
                       xbuf, lnf_s, lnf_b, xnbf);

    // logits = xn @ emb^T + cls_b   (emb is already [V,D] = [N,K])
    launch_gemm(stream, xnbf, embbf, logits, nullptr, cls_b, nullptr,
                BSc, Vc, Dc, Dc, Dc, Vc,
                0, 0, 0, 1, 0, 0, 1, 0, 1.0f, 0, 0, 1);
}
